// SelfAttention_82179904242370
// MI455X (gfx1250) — compile-verified, matrix operand order checked by reading, not run on hardware
//
#include <hip/hip_runtime.h>
#include <hip/hip_bf16.h>

typedef __attribute__((ext_vector_type(16))) _Float16 v16h;
typedef __attribute__((ext_vector_type(8)))  float    v8f;

#define S_LEN   4096
#define D_EMB   512
#define N_HEADS 8
#define D_HEAD  64

// ---------------------------------------------------------------------------
// Fragment loader for 16-bit WMMA operands from a row-major f16 matrix.
// Works for A-fragments (16 rows x 32 K) and for B-fragments when the B matrix
// is available transposed (column n of B == row n of B^T).
// Layout (CDNA5 ISA):
//   lane m (m<16):  row (row0+m), k = kb+{0..7} and kb+{16..23}
//   lane m+16:      row (row0+m), k = kb+{8..15} and kb+{24..31}
// Compiles to two global_load_b128 per fragment.
// ---------------------------------------------------------------------------
__device__ __forceinline__ v16h load_frag16(const _Float16* __restrict__ base,
                                            int ld, int row0, int kb, int lane) {
    const int m = lane & 15, half = lane >> 4;
    const _Float16* p = base + (size_t)(row0 + m) * ld + kb + half * 8;
    v16h f;
#pragma unroll
    for (int i = 0; i < 8; ++i) {
        f[i]     = p[i];
        f[8 + i] = p[16 + i];
    }
    return f;
}

// ---------------------------------------------------------------------------
// Elementwise f32 -> f16 convert
// ---------------------------------------------------------------------------
__global__ __launch_bounds__(256) void cvt_f16_kernel(const float* __restrict__ in,
                                                      _Float16* __restrict__ out, int n) {
    int i = blockIdx.x * 256 + threadIdx.x;
    if (i < n) out[i] = (_Float16)in[i];
}

// f32 [rows, cols] row-major -> f16 [cols, rows] (transpose)
__global__ __launch_bounds__(256) void transpose_f16_kernel(const float* __restrict__ in,
                                                            _Float16* __restrict__ out,
                                                            int rows, int cols) {
    int i = blockIdx.x * 256 + threadIdx.x;
    if (i < rows * cols) {
        int r = i / cols, c = i - r * cols;
        out[(size_t)c * rows + r] = (_Float16)in[i];
    }
}

// ---------------------------------------------------------------------------
// QKV projection: qkv = x @ W_in + b_in, split into per-head Q/K/V (f16).
// Each wave computes a 32(M) x 64(N) tile (each B fragment feeds 2 WMMAs).
// Q is pre-scaled by 1/sqrt(d_head)=0.125 (softmax scale folded in).
// Q,K stored [h][s][64]; V stored TRANSPOSED [h][64][s] so the attention
// kernel's V^T fragments are contiguous b128 loads.
// ---------------------------------------------------------------------------
__global__ __launch_bounds__(256) void qkv_gemm_kernel(const _Float16* __restrict__ x16,
                                                       const _Float16* __restrict__ wt,   // [1536,512] = W_in^T
                                                       const float* __restrict__ b_in,
                                                       _Float16* __restrict__ q,
                                                       _Float16* __restrict__ k,
                                                       _Float16* __restrict__ vt) {
    const int wave = threadIdx.x >> 5, lane = threadIdx.x & 31;
    const int w = blockIdx.x * 8 + wave;          // 3072 waves
    const int TN = (3 * D_EMB) / 64;              // 24 N-tiles
    const int m0 = (w / TN) * 32;
    const int n0 = (w % TN) * 64;

    v8f z = {};
    v8f acc[2][4] = {{z, z, z, z}, {z, z, z, z}};
#pragma unroll 4
    for (int kb = 0; kb < D_EMB; kb += 32) {
        v16h a0 = load_frag16(x16, D_EMB, m0,      kb, lane);
        v16h a1 = load_frag16(x16, D_EMB, m0 + 16, kb, lane);
#pragma unroll
        for (int f = 0; f < 4; ++f) {
            v16h b = load_frag16(wt, D_EMB, n0 + f * 16, kb, lane);
            acc[0][f] = __builtin_amdgcn_wmma_f32_16x16x32_f16(false, a0, false, b,
                                                               (short)0, acc[0][f], false, false);
            acc[1][f] = __builtin_amdgcn_wmma_f32_16x16x32_f16(false, a1, false, b,
                                                               (short)0, acc[1][f], false, false);
        }
    }

    const int half = lane >> 4, nn = lane & 15;
#pragma unroll
    for (int f = 0; f < 4; ++f) {
        const int col = n0 + f * 16 + nn;          // 0..1535
        const int t   = col >> 9;                  // 0=Q 1=K 2=V (uniform per f)
        const int cc  = col & 511;
        const int h   = cc >> 6, d = cc & 63;
        const float scale = (t == 0) ? 0.125f : 1.0f;
        const float bias  = b_in[col];
#pragma unroll
        for (int g = 0; g < 2; ++g)
#pragma unroll
            for (int r = 0; r < 8; ++r) {
                const int row = m0 + g * 16 + 8 * half + r;
                const float val = (acc[g][f][r] + bias) * scale;
                if (t == 2) {
                    // transposed: [h][d][s], contiguous over r
                    vt[(size_t)h * D_HEAD * S_LEN + (size_t)d * S_LEN + row] = (_Float16)val;
                } else {
                    _Float16* dst = (t == 0) ? q : k;
                    dst[(size_t)h * S_LEN * D_HEAD + (size_t)row * D_HEAD + d] = (_Float16)val;
                }
            }
    }
}

// ---------------------------------------------------------------------------
// Flash attention, one wave per (head, 16-query tile).
// Scores computed transposed: S^T = K * Q^T, so the softmax'd probabilities in
// C-layout are *directly* the B-fragment for out^T = V^T * P^T (no LDS, no
// cross-lane transpose). V is stored transposed, so V^T fragments are
// contiguous b128 loads. Online softmax over 32-key blocks.
// ---------------------------------------------------------------------------
__global__ __launch_bounds__(256) void attn_fa_kernel(const _Float16* __restrict__ q,
                                                      const _Float16* __restrict__ k,
                                                      const _Float16* __restrict__ vt,
                                                      _Float16* __restrict__ att,
                                                      const int* __restrict__ cmask) {
    const int causal = *cmask;
    const int wave = threadIdx.x >> 5, lane = threadIdx.x & 31;
    const int w    = blockIdx.x * 8 + wave;       // 2048 waves
    const int head = w >> 8;                      // 256 q-tiles per head
    const int q0   = (w & 255) << 4;
    const size_t HS = (size_t)S_LEN * D_HEAD;
    const _Float16* qh  = q  + head * HS;
    const _Float16* kh  = k  + head * HS;
    const _Float16* vth = vt + head * HS;         // [64][4096]
    const int m = lane & 15, half = lane >> 4;
    const int qabs = q0 + m;                      // this lane's query column

    // Q^T B-fragments (constant across key loop): d-chunks 0..31 and 32..63
    const v16h qb0 = load_frag16(qh, D_HEAD, q0, 0, lane);
    const v16h qb1 = load_frag16(qh, D_HEAD, q0, 32, lane);

    v8f z = {};
    v8f acc[4] = {z, z, z, z};                    // out^T, 4 x (16 d-rows x 16 queries)
    float m_run = -3.0e38f, l_run = 0.0f;

    const int nkb = causal ? ((q0 + 47) >> 5) : (S_LEN / 32);
    for (int kb32 = 0; kb32 < nkb; ++kb32) {
        const int kb = kb32 << 5;

        // prefetch next key block (32 K rows; 64 V^T row segments)
        if (kb32 + 1 < nkb) {
            __builtin_prefetch(kh + (size_t)(kb + 32 + lane) * D_HEAD, 0, 1);
            __builtin_prefetch(vth + (size_t)lane * S_LEN + kb + 32, 0, 1);
            __builtin_prefetch(vth + (size_t)(lane + 32) * S_LEN + kb + 32, 0, 1);
        }

        // --- scores: two 16-key tiles of S^T = K * Q^T (d accumulated 32+32)
        v8f s0 = z, s1 = z;
        {
            v16h ka0 = load_frag16(kh, D_HEAD, kb, 0, lane);
            v16h ka1 = load_frag16(kh, D_HEAD, kb, 32, lane);
            s0 = __builtin_amdgcn_wmma_f32_16x16x32_f16(false, ka0, false, qb0, (short)0, s0, false, false);
            s0 = __builtin_amdgcn_wmma_f32_16x16x32_f16(false, ka1, false, qb1, (short)0, s0, false, false);
            v16h kc0 = load_frag16(kh, D_HEAD, kb + 16, 0, lane);
            v16h kc1 = load_frag16(kh, D_HEAD, kb + 16, 32, lane);
            s1 = __builtin_amdgcn_wmma_f32_16x16x32_f16(false, kc0, false, qb0, (short)0, s1, false, false);
            s1 = __builtin_amdgcn_wmma_f32_16x16x32_f16(false, kc1, false, qb1, (short)0, s1, false, false);
        }

        // --- causal mask + row max (row = query, spread over lane & lane^16)
        float rmax = -3.0e38f;
#pragma unroll
        for (int r = 0; r < 8; ++r) {
            const int key0 = kb + 8 * half + r;   // key index of s0[r]
            float a = s0[r];
            float b = s1[r];
            if (causal && key0 > qabs)      a = -3.0e38f;
            if (causal && key0 + 16 > qabs) b = -3.0e38f;
            s0[r] = a; s1[r] = b;
            rmax = fmaxf(rmax, fmaxf(a, b));
        }
        rmax = fmaxf(rmax, __shfl_xor(rmax, 16, 32));
        const float m_new = fmaxf(m_run, rmax);
        const float fac   = __expf(m_run - m_new);

        // --- exp + sum; pack P^T directly into B-fragment layout
        float lsum = 0.0f;
        v16h pb;
#pragma unroll
        for (int r = 0; r < 8; ++r) {
            const float e0 = __expf(s0[r] - m_new);
            const float e1 = __expf(s1[r] - m_new);
            lsum += e0 + e1;
            pb[r]     = (_Float16)e0;             // keys kb+{0..7 | 8..15}
            pb[8 + r] = (_Float16)e1;             // keys kb+{16..23 | 24..31}
        }
        lsum += __shfl_xor(lsum, 16, 32);
        l_run = l_run * fac + lsum;
        m_run = m_new;

        // --- rescale running output
#pragma unroll
        for (int f = 0; f < 4; ++f)
#pragma unroll
            for (int r = 0; r < 8; ++r) acc[f][r] *= fac;

        // --- out^T += V^T(d x 32keys) * P^T(32keys x 16q); contiguous loads
#pragma unroll
        for (int f = 0; f < 4; ++f) {
            v16h vf = load_frag16(vth, S_LEN, f * 16, kb, lane);
            acc[f] = __builtin_amdgcn_wmma_f32_16x16x32_f16(false, vf, false, pb,
                                                            (short)0, acc[f], false, false);
        }
    }

    // --- normalize and store to head-concatenated [S, 512] f16
    const float inv_l = 1.0f / l_run;
    _Float16* orow = att + (size_t)(q0 + m) * D_EMB + head * D_HEAD;
#pragma unroll
    for (int f = 0; f < 4; ++f)
#pragma unroll
        for (int r = 0; r < 8; ++r)
            orow[f * 16 + 8 * half + r] = (_Float16)(acc[f][r] * inv_l);
}

// ---------------------------------------------------------------------------
// Output projection: out = attn @ W_out + b_out  (f32 output)
// 32(M) x 64(N) tile per wave.
// ---------------------------------------------------------------------------
__global__ __launch_bounds__(256) void out_gemm_kernel(const _Float16* __restrict__ attn,
                                                       const _Float16* __restrict__ wt,   // [512,512] = W_out^T
                                                       const float* __restrict__ b_out,
                                                       float* __restrict__ out) {
    const int wave = threadIdx.x >> 5, lane = threadIdx.x & 31;
    const int w = blockIdx.x * 8 + wave;          // 1024 waves
    const int TN = D_EMB / 64;                    // 8
    const int m0 = (w / TN) * 32;
    const int n0 = (w % TN) * 64;

    v8f z = {};
    v8f acc[2][4] = {{z, z, z, z}, {z, z, z, z}};
#pragma unroll 4
    for (int kb = 0; kb < D_EMB; kb += 32) {
        v16h a0 = load_frag16(attn, D_EMB, m0,      kb, lane);
        v16h a1 = load_frag16(attn, D_EMB, m0 + 16, kb, lane);
#pragma unroll
        for (int f = 0; f < 4; ++f) {
            v16h b = load_frag16(wt, D_EMB, n0 + f * 16, kb, lane);
            acc[0][f] = __builtin_amdgcn_wmma_f32_16x16x32_f16(false, a0, false, b,
                                                               (short)0, acc[0][f], false, false);
            acc[1][f] = __builtin_amdgcn_wmma_f32_16x16x32_f16(false, a1, false, b,
                                                               (short)0, acc[1][f], false, false);
        }
    }

    const int half = lane >> 4, nn = lane & 15;
#pragma unroll
    for (int f = 0; f < 4; ++f) {
        const int col = n0 + f * 16 + nn;
        const float bias = b_out[col];
#pragma unroll
        for (int g = 0; g < 2; ++g)
#pragma unroll
            for (int r = 0; r < 8; ++r) {
                const int row = m0 + g * 16 + 8 * half + r;
                out[(size_t)row * D_EMB + col] = acc[g][f][r] + bias;
            }
    }
}

// ---------------------------------------------------------------------------
extern "C" void kernel_launch(void* const* d_in, const int* in_sizes, int n_in,
                              void* d_out, int out_size, void* d_ws, size_t ws_size,
                              hipStream_t stream) {
    const float* x     = (const float*)d_in[0];
    const float* W_in  = (const float*)d_in[1];
    const float* b_in  = (const float*)d_in[2];
    const float* W_out = (const float*)d_in[3];
    const float* b_out = (const float*)d_in[4];
    const int*   cmask = (const int*)d_in[5];

    const size_t MB = 1u << 20;
    char* ws = (char*)d_ws;
    _Float16* x16    = (_Float16*)(ws + 0);                    // 4096x512   (4 MiB)
    _Float16* wint   = (_Float16*)(ws + 4 * MB);               // 1536x512   (1.5 MiB)
    _Float16* woutt  = (_Float16*)(ws + 4 * MB + MB / 2 + MB); // 512x512    (0.5 MiB)
    _Float16* q16    = (_Float16*)(ws + 6 * MB);               // [8][4096][64]
    _Float16* k16    = (_Float16*)(ws + 10 * MB);              // [8][4096][64]
    _Float16* v16t   = (_Float16*)(ws + 14 * MB);              // [8][64][4096] (transposed)
    _Float16* att16  = (_Float16*)(ws + 18 * MB);              // [4096][512]

    // 1) x -> f16
    cvt_f16_kernel<<<(S_LEN * D_EMB) / 256, 256, 0, stream>>>(x, x16, S_LEN * D_EMB);
    // 2) W_in^T, W_out^T -> f16
    transpose_f16_kernel<<<(D_EMB * 3 * D_EMB) / 256, 256, 0, stream>>>(W_in, wint, D_EMB, 3 * D_EMB);
    transpose_f16_kernel<<<(D_EMB * D_EMB) / 256, 256, 0, stream>>>(W_out, woutt, D_EMB, D_EMB);
    // 3) QKV projection (3072 waves -> 384 blocks of 8 waves)
    qkv_gemm_kernel<<<384, 256, 0, stream>>>(x16, wint, b_in, q16, k16, v16t);
    // 4) flash attention (2048 waves -> 256 blocks)
    attn_fa_kernel<<<256, 256, 0, stream>>>(q16, k16, v16t, att16, cmask);
    // 5) output projection (1024 waves -> 128 blocks)
    out_gemm_kernel<<<128, 256, 0, stream>>>(att16, woutt, b_out, (float*)d_out);
}
